// TimesFMAttention_47493748359871
// MI455X (gfx1250) — compile-verified, matrix operand order checked by reading, not run on hardware
//
#include <hip/hip_runtime.h>
#include <hip/hip_bf16.h>

#define HIDDEN   1280
#define NUM_HEADS 16
#define NUM_KV    4
#define HEAD_DIM  80
#define DPAD      96            // head dim padded to 3*32 for K-dim of WMMA
#define Q_SIZE   1280
#define KV_SIZE  320
#define QKV_OUT  1920
#define SEQ      2048
#define BATCH    4
#define MTOT     (BATCH*SEQ)    // 8192

#define TILE_AROWS 64
#define TILE_BROWS 128
#define LROW 40                 // GEMM LDS row stride (elems): conflict-free reads
#define KROW 104                // attn K-tile row stride (32 keys x 96 d)
#define VROW 40                 // attn V-tile row stride (80 d x 32 keys)

#ifndef __has_builtin
#define __has_builtin(x) 0
#endif
#if __has_builtin(__builtin_amdgcn_global_load_async_to_lds_b128)
#define HAS_ASYNC 1
#else
#define HAS_ASYNC 0
#endif

typedef __attribute__((ext_vector_type(16))) __bf16 v16bf;
typedef __attribute__((ext_vector_type(8)))  __bf16 v8bf;
typedef __attribute__((ext_vector_type(8)))  float  v8f;
typedef __attribute__((ext_vector_type(4)))  float  v4f;

#if HAS_ASYNC
typedef int v4i_vs __attribute__((__vector_size__(16)));
typedef __attribute__((address_space(1))) v4i_vs glob_v4i;
typedef __attribute__((address_space(3))) v4i_vs lds_v4i;

__device__ inline glob_v4i* as_global_v4i(const void* p) {
    return (glob_v4i*)(unsigned long long)(uintptr_t)p;
}
__device__ inline lds_v4i* as_lds_v4i(void* p) {
    return (lds_v4i*)(unsigned int)(uintptr_t)p;
}
#endif

// One 16B async (or sync) copy global -> LDS
__device__ inline void copy16(const __bf16* g, __bf16* l) {
#if HAS_ASYNC
    __builtin_amdgcn_global_load_async_to_lds_b128(as_global_v4i(g), as_lds_v4i(l), 0, 0);
#else
    *(v8bf*)l = *(const v8bf*)g;
#endif
}

// Load a 16x32 (A) or 32x16 (B) bf16 fragment half-pair: 8 elems at p, 8 at p+16.
__device__ inline v16bf load_frag(const __bf16* p) {
    v8bf lo = *(const v8bf*)p;
    v8bf hi = *(const v8bf*)(p + 16);
    v16bf r;
#pragma unroll
    for (int i = 0; i < 8; ++i) { r[i] = lo[i]; r[i + 8] = hi[i]; }
    return r;
}

__device__ inline v8f wmma_bf16(v16bf a, v16bf b, v8f c) {
    return __builtin_amdgcn_wmma_f32_16x16x32_bf16(false, a, false, b, (short)0, c,
                                                   false, false);
}

// ---------------- conversion kernels ----------------
__global__ void tfm_cvt_bf16(const float* __restrict__ src, __bf16* __restrict__ dst,
                             int n8) {
    int i = blockIdx.x * blockDim.x + threadIdx.x;
    int stride = gridDim.x * blockDim.x;
    for (; i < n8; i += stride) {
        v4f lo = *(const v4f*)(src + i * 8);
        v4f hi = *(const v4f*)(src + i * 8 + 4);
        v8bf o;
#pragma unroll
        for (int j = 0; j < 4; ++j) { o[j] = (__bf16)lo[j]; o[j + 4] = (__bf16)hi[j]; }
        *(v8bf*)(dst + i * 8) = o;
    }
}

__global__ void tfm_scale(const float* __restrict__ sc, float* __restrict__ out) {
    int d = threadIdx.x;
    if (d < HEAD_DIM) {
        float x = sc[d];
        float sp = (x > 20.f) ? x : log1pf(__expf(x));
        out[d] = 1.442695041f * rsqrtf((float)HEAD_DIM) * sp;
    }
}

// ---------------- cooperative tile staging (global -> LDS, double buffered) --------
struct Stage { v8bf a, b0, b1; };

__device__ inline void stage_issue(const __bf16* __restrict__ Ag,
                                   const __bf16* __restrict__ Wg, int k, int tid,
                                   __bf16* Al, __bf16* Bl, Stage& st) {
    const int r = tid >> 2, c = (tid & 3) * 8;
#if HAS_ASYNC
    copy16(Ag + r * HIDDEN + k + c, Al + r * LROW + c);
    copy16(Wg + r * HIDDEN + k + c, Bl + r * LROW + c);
    copy16(Wg + (r + 64) * HIDDEN + k + c, Bl + (r + 64) * LROW + c);
#else
    st.a  = *(const v8bf*)(Ag + r * HIDDEN + k + c);
    st.b0 = *(const v8bf*)(Wg + r * HIDDEN + k + c);
    st.b1 = *(const v8bf*)(Wg + (r + 64) * HIDDEN + k + c);
#endif
}

__device__ inline void stage_commit(int tid, __bf16* Al, __bf16* Bl, Stage& st) {
#if HAS_ASYNC
    asm volatile("s_wait_asynccnt 0x0" ::: "memory");
#else
    const int r = tid >> 2, c = (tid & 3) * 8;
    *(v8bf*)(Al + r * LROW + c)        = st.a;
    *(v8bf*)(Bl + r * LROW + c)        = st.b0;
    *(v8bf*)(Bl + (r + 64) * LROW + c) = st.b1;
#endif
    __syncthreads();
}

// ---------------- QKV projection GEMM ----------------
__global__ __launch_bounds__(256, 2) void tfm_qkv_gemm(
    const __bf16* __restrict__ A, const __bf16* __restrict__ W,
    const float* __restrict__ bias, const float* __restrict__ scale,
    __bf16* __restrict__ qbuf, __bf16* __restrict__ kbuf, __bf16* __restrict__ vtbuf) {
    __shared__ __align__(16) __bf16 At[2][TILE_AROWS][LROW];
    __shared__ __align__(16) __bf16 Bt[2][TILE_BROWS][LROW];
    const int tid = threadIdx.x;
    const int lane = tid & 31, lane_lo = lane & 15, lane_hi = lane >> 4;
    const int w = tid >> 5, wm = w & 3, wn = w >> 2;
    const int row0 = blockIdx.y * 64 + wm * 16;
    const int col0 = blockIdx.x * 128 + wn * 64;
    const __bf16* Ag = A + (size_t)blockIdx.y * 64 * HIDDEN;
    const __bf16* Wg = W + (size_t)blockIdx.x * 128 * HIDDEN;

    v8f acc[4] = {v8f{}, v8f{}, v8f{}, v8f{}};
    Stage st;
    stage_issue(Ag, Wg, 0, tid, &At[0][0][0], &Bt[0][0][0], st);
    stage_commit(tid, &At[0][0][0], &Bt[0][0][0], st);
    for (int k = 0; k < HIDDEN; k += 32) {
        const int buf = (k >> 5) & 1;
        if (k + 32 < HIDDEN)
            stage_issue(Ag, Wg, k + 32, tid, &At[buf ^ 1][0][0], &Bt[buf ^ 1][0][0], st);
        v16bf af = load_frag(&At[buf][wm * 16 + lane_lo][lane_hi * 8]);
#pragma unroll
        for (int j = 0; j < 4; ++j) {
            v16bf bf = load_frag(&Bt[buf][wn * 64 + j * 16 + lane_lo][lane_hi * 8]);
            acc[j] = wmma_bf16(af, bf, acc[j]);
        }
        if (k + 32 < HIDDEN)
            stage_commit(tid, &At[buf ^ 1][0][0], &Bt[buf ^ 1][0][0], st);
    }

    const int b  = row0 >> 11;
    const int s0 = (row0 & (SEQ - 1)) + 8 * lane_hi;
#pragma unroll
    for (int j = 0; j < 4; ++j) {
        int n = col0 + j * 16 + lane_lo;
        float bv = bias[n];
        if (n < Q_SIZE) {
            int h = n / HEAD_DIM, d = n % HEAD_DIM;
            float scv = scale[d];
            __bf16* o = qbuf + (((b * NUM_HEADS + h) * SEQ) + s0) * DPAD + d;
#pragma unroll
            for (int v = 0; v < 8; ++v) o[v * DPAD] = (__bf16)((acc[j][v] + bv) * scv);
        } else if (n < Q_SIZE + KV_SIZE) {
            int nk = n - Q_SIZE, kvh = nk / HEAD_DIM, d = nk % HEAD_DIM;
            __bf16* o = kbuf + (((b * NUM_KV + kvh) * SEQ) + s0) * DPAD + d;
#pragma unroll
            for (int v = 0; v < 8; ++v) o[v * DPAD] = (__bf16)(acc[j][v] + bv);
        } else {
            int nv = n - Q_SIZE - KV_SIZE, kvh = nv / HEAD_DIM, d = nv % HEAD_DIM;
            v8bf pk;
#pragma unroll
            for (int v = 0; v < 8; ++v) pk[v] = (__bf16)(acc[j][v] + bv);
            *(v8bf*)(vtbuf + ((b * NUM_KV + kvh) * HEAD_DIM + d) * SEQ + s0) = pk;
        }
    }
}

// ---------------- flash attention (cooperative async K/V staging) ----------------
// grid.x = SEQ/128 q-tiles, grid.y = B*H ; each wave: 16 q rows, full d.
// Per 32-key chunk the block stages K (32x96) and V^T (80x32) tiles in LDS once,
// double buffered via the async-to-LDS path; all 8 waves share them.
__global__ __launch_bounds__(256, 2) void tfm_attn(
    const __bf16* __restrict__ qbuf, const __bf16* __restrict__ kbuf,
    const __bf16* __restrict__ vtbuf, __bf16* __restrict__ aout) {
    __shared__ __align__(16) __bf16 Kt[2][32][KROW];
    __shared__ __align__(16) __bf16 Vt[2][80][VROW];
    __shared__ __align__(16) __bf16 pl[8][16][40];  // per-wave P staging
    const int tid = threadIdx.x;
    const int lane = tid & 31, lane_lo = lane & 15, lane_hi = lane >> 4;
    const int w = tid >> 5;
    const int bh = blockIdx.y, b = bh >> 4, h = bh & 15, kvh = h >> 2;
    const int q0 = blockIdx.x * 128 + w * 16;

    const __bf16* qp = qbuf + ((size_t)bh * SEQ + q0 + lane_lo) * DPAD + lane_hi * 8;
    v16bf qfrag[3];
#pragma unroll
    for (int ks = 0; ks < 3; ++ks) qfrag[ks] = load_frag(qp + ks * 32);

    const __bf16* kp = kbuf + (size_t)(b * NUM_KV + kvh) * SEQ * DPAD;
    const __bf16* vp = vtbuf + (size_t)(b * NUM_KV + kvh) * HEAD_DIM * SEQ;

    v8f o[5] = {v8f{}, v8f{}, v8f{}, v8f{}, v8f{}};
    float mrow[8], lrow[8];
#pragma unroll
    for (int v = 0; v < 8; ++v) { mrow[v] = -1e30f; lrow[v] = 0.f; }
    __bf16* myp = &pl[w][0][0];

    // stage K (32 rows x 96 d = 384 16B chunks) + V (80 rows x 32 keys = 320 chunks)
    auto issue_kv = [&](int kc, int buf) {
        {
            int row = tid / 12, cc = (tid % 12) * 8;
            copy16(kp + (kc + row) * DPAD + cc, &Kt[buf][row][cc]);
            if (tid < 128) {
                int id = tid + 256;
                row = id / 12; cc = (id % 12) * 8;
                copy16(kp + (kc + row) * DPAD + cc, &Kt[buf][row][cc]);
            }
        }
        {
            int row = tid >> 2, cc = (tid & 3) * 8;
            copy16(vp + (size_t)row * SEQ + kc + cc, &Vt[buf][row][cc]);
            if (tid < 64) {
                int id = tid + 256;
                row = id >> 2; cc = (id & 3) * 8;
                copy16(vp + (size_t)row * SEQ + kc + cc, &Vt[buf][row][cc]);
            }
        }
    };
    auto commit_kv = [&]() {
#if HAS_ASYNC
        asm volatile("s_wait_asynccnt 0x0" ::: "memory");
#endif
        __syncthreads();
    };

    auto chunk = [&](int buf) {
        v8f stc[2] = {v8f{}, v8f{}};
#pragma unroll
        for (int t = 0; t < 2; ++t) {
            const __bf16* kbase = &Kt[buf][t * 16 + lane_lo][lane_hi * 8];
#pragma unroll
            for (int ks = 0; ks < 3; ++ks) {
                v16bf kf = load_frag(kbase + ks * 32);
                stc[t] = wmma_bf16(qfrag[ks], kf, stc[t]);
            }
        }
        v16bf vf[5];
#pragma unroll
        for (int dt = 0; dt < 5; ++dt)
            vf[dt] = load_frag(&Vt[buf][dt * 16 + lane_lo][lane_hi * 8]);
        float pr0[8], pr1[8], fac[8];
#pragma unroll
        for (int v = 0; v < 8; ++v) {
            float cm = fmaxf(stc[0][v], stc[1][v]);
            cm = fmaxf(cm, __shfl_xor(cm, 1, 32));
            cm = fmaxf(cm, __shfl_xor(cm, 2, 32));
            cm = fmaxf(cm, __shfl_xor(cm, 4, 32));
            cm = fmaxf(cm, __shfl_xor(cm, 8, 32));
            float mn = fmaxf(mrow[v], cm);
            fac[v] = __expf(mrow[v] - mn);
            mrow[v] = mn;
            pr0[v] = __expf(stc[0][v] - mn);
            pr1[v] = __expf(stc[1][v] - mn);
            float rs = pr0[v] + pr1[v];
            rs += __shfl_xor(rs, 1, 32);
            rs += __shfl_xor(rs, 2, 32);
            rs += __shfl_xor(rs, 4, 32);
            rs += __shfl_xor(rs, 8, 32);
            lrow[v] = lrow[v] * fac[v] + rs;
        }
#pragma unroll
        for (int dt = 0; dt < 5; ++dt)
#pragma unroll
            for (int v = 0; v < 8; ++v) o[dt][v] *= fac[v];
        // scatter P (C layout) -> LDS, reload as A fragment
#pragma unroll
        for (int v = 0; v < 8; ++v) {
            int m = v + 8 * lane_hi;
            myp[m * 40 + lane_lo]      = (__bf16)pr0[v];
            myp[m * 40 + 16 + lane_lo] = (__bf16)pr1[v];
        }
        asm volatile("s_wait_dscnt 0x0" ::: "memory");
        v16bf pf;
        {
            const __bf16* pr = myp + lane_lo * 40 + lane_hi * 8;
            v8bf plo = *(const v8bf*)pr;
            v8bf phi = *(const v8bf*)(pr + 16);
#pragma unroll
            for (int i = 0; i < 8; ++i) { pf[i] = plo[i]; pf[i + 8] = phi[i]; }
        }
#pragma unroll
        for (int dt = 0; dt < 5; ++dt) o[dt] = wmma_bf16(pf, vf[dt], o[dt]);
    };

    issue_kv(0, 0);
    commit_kv();
    for (int kc = 0; kc < SEQ; kc += 32) {
        const int buf = (kc >> 5) & 1;
        if (kc + 32 < SEQ) issue_kv(kc + 32, buf ^ 1);
        chunk(buf);
        if (kc + 32 < SEQ) commit_kv();
    }

#pragma unroll
    for (int dt = 0; dt < 5; ++dt)
#pragma unroll
        for (int v = 0; v < 8; ++v) {
            int m = v + 8 * lane_hi;
            int row = b * SEQ + q0 + m;
            aout[(size_t)row * HIDDEN + h * HEAD_DIM + dt * 16 + lane_lo] =
                (__bf16)(o[dt][v] / lrow[v]);
        }
}

// ---------------- output projection GEMM ----------------
__global__ __launch_bounds__(256, 2) void tfm_oproj(
    const __bf16* __restrict__ A, const __bf16* __restrict__ W,
    const float* __restrict__ bias, float* __restrict__ out) {
    __shared__ __align__(16) __bf16 At[2][TILE_AROWS][LROW];
    __shared__ __align__(16) __bf16 Bt[2][TILE_BROWS][LROW];
    const int tid = threadIdx.x;
    const int lane = tid & 31, lane_lo = lane & 15, lane_hi = lane >> 4;
    const int w = tid >> 5, wm = w & 3, wn = w >> 2;
    const int row0 = blockIdx.y * 64 + wm * 16;
    const int col0 = blockIdx.x * 128 + wn * 64;
    const __bf16* Ag = A + (size_t)blockIdx.y * 64 * HIDDEN;
    const __bf16* Wg = W + (size_t)blockIdx.x * 128 * HIDDEN;

    v8f acc[4] = {v8f{}, v8f{}, v8f{}, v8f{}};
    Stage st;
    stage_issue(Ag, Wg, 0, tid, &At[0][0][0], &Bt[0][0][0], st);
    stage_commit(tid, &At[0][0][0], &Bt[0][0][0], st);
    for (int k = 0; k < HIDDEN; k += 32) {
        const int buf = (k >> 5) & 1;
        if (k + 32 < HIDDEN)
            stage_issue(Ag, Wg, k + 32, tid, &At[buf ^ 1][0][0], &Bt[buf ^ 1][0][0], st);
        v16bf af = load_frag(&At[buf][wm * 16 + lane_lo][lane_hi * 8]);
#pragma unroll
        for (int j = 0; j < 4; ++j) {
            v16bf bf = load_frag(&Bt[buf][wn * 64 + j * 16 + lane_lo][lane_hi * 8]);
            acc[j] = wmma_bf16(af, bf, acc[j]);
        }
        if (k + 32 < HIDDEN)
            stage_commit(tid, &At[buf ^ 1][0][0], &Bt[buf ^ 1][0][0], st);
    }

#pragma unroll
    for (int j = 0; j < 4; ++j) {
        int n = col0 + j * 16 + lane_lo;
        float bv = bias[n];
        float* op = out + (size_t)(row0 + 8 * lane_hi) * HIDDEN + n;
#pragma unroll
        for (int v = 0; v < 8; ++v) op[(size_t)v * HIDDEN] = acc[j][v] + bv;
    }
}

extern "C" void kernel_launch(void* const* d_in, const int* in_sizes, int n_in,
                              void* d_out, int out_size, void* d_ws, size_t ws_size,
                              hipStream_t stream) {
    const float* hs   = (const float*)d_in[0];
    const float* scl  = (const float*)d_in[1];
    const float* qkvw = (const float*)d_in[2];
    const float* qkvb = (const float*)d_in[3];
    const float* ow   = (const float*)d_in[4];
    const float* ob   = (const float*)d_in[5];
    float* out = (float*)d_out;

    char* base = (char*)d_ws;
    size_t off = 0;
    auto alloc = [&](size_t bytes) {
        void* p = base + off;
        off = (off + bytes + 255) & ~(size_t)255;
        return p;
    };
    const size_t hsb_b = (size_t)MTOT * HIDDEN * 2;
    const size_t qwb_b = (size_t)QKV_OUT * HIDDEN * 2;
    const size_t owb_b = (size_t)HIDDEN * HIDDEN * 2;
    const size_t qb_b  = (size_t)BATCH * NUM_HEADS * SEQ * DPAD * 2;
    const size_t kb_b  = (size_t)BATCH * NUM_KV * SEQ * DPAD * 2;
    const size_t vt_b  = (size_t)BATCH * NUM_KV * HEAD_DIM * SEQ * 2;
    const size_t ao_b  = (size_t)MTOT * HIDDEN * 2;

    __bf16* hsb = (__bf16*)alloc(hsb_b);
    __bf16* qwb = (__bf16*)alloc(qwb_b);
    __bf16* owb = (__bf16*)alloc(owb_b);
    __bf16* qb  = (__bf16*)alloc(qb_b);
    __bf16* kb  = (__bf16*)alloc(kb_b);
    __bf16* vt  = (__bf16*)alloc(vt_b);
    __bf16* ao  = (__bf16*)alloc(ao_b);
    float* scb  = (float*)alloc(HEAD_DIM * sizeof(float));
    if (off > ws_size) return;

    tfm_cvt_bf16<<<2048, 256, 0, stream>>>(hs, hsb, MTOT * HIDDEN / 8);
    tfm_cvt_bf16<<<1024, 256, 0, stream>>>(qkvw, qwb, QKV_OUT * HIDDEN / 8);
    tfm_cvt_bf16<<<1024, 256, 0, stream>>>(ow, owb, HIDDEN * HIDDEN / 8);
    tfm_scale<<<1, 128, 0, stream>>>(scl, scb);
    (void)hipMemsetAsync(qb, 0, qb_b, stream);   // zero d-padding (80..95)
    (void)hipMemsetAsync(kb, 0, kb_b, stream);

    tfm_qkv_gemm<<<dim3(QKV_OUT / 128, MTOT / 64), 256, 0, stream>>>(
        hsb, qwb, qkvb, scb, qb, kb, vt);
    tfm_attn<<<dim3(SEQ / 128, BATCH * NUM_HEADS), 256, 0, stream>>>(qb, kb, vt, ao);
    tfm_oproj<<<dim3(HIDDEN / 128, MTOT / 64), 256, 0, stream>>>(ao, owb, ob, out);
}
